// SelfAttention_20409684590630
// MI455X (gfx1250) — matrix operand order repaired and checked by reading, not compiled, not run on hardware
//
#include <hip/hip_runtime.h>

typedef __attribute__((ext_vector_type(16))) __bf16 v16bf;
typedef __attribute__((ext_vector_type(8)))  __bf16 v8bf;
typedef __attribute__((ext_vector_type(8)))  float  v8f;

#define DIM 1024
#define NH 16
#define HD 64
#define B_ 8
#define S_ 1025
#define M_TOT (B_ * S_)   /* 8200 */
#define M_PAD 8224        /* 257 * 32 */
#define M_TILES2 257      /* 32-row tiles */
#define SPAD 1088         /* 34 * 32: key-padded sequence (+1 tile for prefetch) */
#define QTILES 65

// ---- helpers ---------------------------------------------------------------

// A-frag (16x32 bf16) per-lane load, contiguous k order:
// lane L: M = L&15; Kbase = 16*(L>>4); elems 0..15 = K base..base+15.
// Caller passes row ptr already offset by Kbase.
static __device__ inline v16bf load_a16(const __bf16* p) {
  v8bf lo = *(const v8bf*)(p);
  v8bf hi = *(const v8bf*)(p + 8);
  v16bf a;
#pragma unroll
  for (int i = 0; i < 8; ++i) { a[i] = lo[i]; a[i + 8] = hi[i]; }
  return a;
}

static __device__ inline v8f wmma_bf16(v16bf a, v16bf b, v8f c) {
  return __builtin_amdgcn_wmma_f32_16x16x32_bf16(false, a, false, b, (short)0, c,
                                                 false, false);
}

// ---- kernel 1: fp32 -> bf16 conversion ------------------------------------
__global__ __launch_bounds__(256) void cvt_bf16(const float* __restrict__ src,
                                                __bf16* __restrict__ dst, int n) {
  int i = (blockIdx.x * 256 + threadIdx.x) * 4;
  if (i < n) {
    float4 f = *(const float4*)(src + i);
    dst[i + 0] = (__bf16)f.x;
    dst[i + 1] = (__bf16)f.y;
    dst[i + 2] = (__bf16)f.z;
    dst[i + 3] = (__bf16)f.w;
  }
}

// ---- kernel 2: LayerNorm (fp32 in, bf16 out) -------------------------------
__global__ __launch_bounds__(256) void ln_kernel(const float* __restrict__ x,
                                                 const float* __restrict__ w,
                                                 const float* __restrict__ b,
                                                 __bf16* __restrict__ h) {
  int row = blockIdx.x;
  int t = threadIdx.x;
  const float4* xv = (const float4*)(x + (size_t)row * DIM);
  float4 v = xv[t];
  float s  = v.x + v.y + v.z + v.w;
  float ss = v.x * v.x + v.y * v.y + v.z * v.z + v.w * v.w;
#pragma unroll
  for (int m = 16; m; m >>= 1) { s += __shfl_xor(s, m); ss += __shfl_xor(ss, m); }
  __shared__ float rs[8], rq[8];
  if ((t & 31) == 0) { rs[t >> 5] = s; rq[t >> 5] = ss; }
  __syncthreads();
  s = 0.f; ss = 0.f;
#pragma unroll
  for (int i = 0; i < 8; ++i) { s += rs[i]; ss += rq[i]; }
  float mu = s * (1.f / DIM);
  float var = ss * (1.f / DIM) - mu * mu;
  float rstd = rsqrtf(var + 1e-5f);
  float4 wv = ((const float4*)w)[t];
  float4 bv = ((const float4*)b)[t];
  __bf16* hr = h + (size_t)row * DIM + t * 4;
  hr[0] = (__bf16)((v.x - mu) * rstd * wv.x + bv.x);
  hr[1] = (__bf16)((v.y - mu) * rstd * wv.y + bv.y);
  hr[2] = (__bf16)((v.z - mu) * rstd * wv.z + bv.z);
  hr[3] = (__bf16)((v.w - mu) * rstd * wv.w + bv.w);
}

// ---- kernel 3: QKV GEMM + bias + RoPE + scatter ----------------------------
// Each wave: 32 rows (tokens) x 64 cols (one head's q, k, or v slice).
// Software-pipelined: frags for K-step k+1 are in flight under step k's WMMAs;
// sched_barrier pins the prefetch clause above the WMMA block.
__global__ __launch_bounds__(256) void qkv_gemm(
    const __bf16* __restrict__ h, const __bf16* __restrict__ w,
    const float* __restrict__ bias, const int* __restrict__ pos,
    const float* __restrict__ theta, __bf16* __restrict__ qb,
    __bf16* __restrict__ kb, __bf16* __restrict__ vt) {
  int lane = threadIdx.x & 31, wid = threadIdx.x >> 5;
  int mt = blockIdx.x;
  int cg = blockIdx.y * 8 + wid;  // 0..47 -> 64-col group
  int n0 = cg * 64;
  int lh = lane & 15, lz = lane >> 4;

  const __bf16* ap0 = h + (size_t)(mt * 32 + lh) * DIM + lz * 16;
  const __bf16* ap1 = ap0 + (size_t)16 * DIM;
  const __bf16* bp  = w + (size_t)(n0 + lh) * DIM + lz * 16;

  v8f acc[8] = {{}, {}, {}, {}, {}, {}, {}, {}};
  v16bf a0 = load_a16(ap0), a1 = load_a16(ap1);
  v16bf bf[4];
#pragma unroll
  for (int t = 0; t < 4; ++t) bf[t] = *(const v16bf*)(bp + (size_t)t * 16 * DIM);

#pragma unroll 4
  for (int k0 = 32; k0 < DIM; k0 += 32) {
    v16bf na0 = load_a16(ap0 + k0);
    v16bf na1 = load_a16(ap1 + k0);
    v16bf nb[4];
#pragma unroll
    for (int t = 0; t < 4; ++t)
      nb[t] = *(const v16bf*)(bp + (size_t)t * 16 * DIM + k0);
    __builtin_amdgcn_sched_barrier(0);  // keep prefetch ahead of the WMMAs
#pragma unroll
    for (int t = 0; t < 4; ++t) {
      acc[t]     = wmma_bf16(a0, bf[t], acc[t]);
      acc[4 + t] = wmma_bf16(a1, bf[t], acc[4 + t]);
    }
    a0 = na0; a1 = na1;
#pragma unroll
    for (int t = 0; t < 4; ++t) bf[t] = nb[t];
  }
#pragma unroll
  for (int t = 0; t < 4; ++t) {
    acc[t]     = wmma_bf16(a0, bf[t], acc[t]);
    acc[4 + t] = wmma_bf16(a1, bf[t], acc[4 + t]);
  }

  float bs[4];
#pragma unroll
  for (int t = 0; t < 4; ++t) bs[t] = bias[n0 + t * 16 + lh];
  int sel = n0 / DIM;             // 0=q, 1=k, 2=v
  int head = (n0 % DIM) / HD;

#pragma unroll
  for (int g = 0; g < 2; ++g) {
#pragma unroll
    for (int r = 0; r < 8; ++r) {
      int m = mt * 32 + g * 16 + r + lz * 8;
      if (m >= M_TOT) continue;
      int bb = m / S_;
      int ss = m - bb * S_;
      size_t rowbase = ((size_t)(bb * NH + head) * SPAD + ss) * HD;
      float c0 = acc[g * 4 + 0][r] + bs[0];
      float c1 = acc[g * 4 + 1][r] + bs[1];
      float c2 = acc[g * 4 + 2][r] + bs[2];
      float c3 = acc[g * 4 + 3][r] + bs[3];
      if (sel == 2) {
        size_t tb = (size_t)(bb * NH + head) * HD;  // transposed V: [bh][d][SPAD]
        vt[(tb + 0  + lh) * SPAD + ss] = (__bf16)c0;
        vt[(tb + 16 + lh) * SPAD + ss] = (__bf16)c1;
        vt[(tb + 32 + lh) * SPAD + ss] = (__bf16)c2;
        vt[(tb + 48 + lh) * SPAD + ss] = (__bf16)c3;
      } else {
        __bf16* dst = (sel == 0) ? qb : kb;
        int p = pos[m];
        const float* thr = theta + ((size_t)p * NH + head) * 32;
        float th0 = thr[lh], th1 = thr[16 + lh];
        float cc0 = __cosf(th0), sn0 = __sinf(th0);
        float cc1 = __cosf(th1), sn1 = __sinf(th1);
        // pair (j, j+32): x1 = tiles 0,1 ; x2 = tiles 2,3 (same lane!)
        dst[rowbase + 0  + lh] = (__bf16)(c0 * cc0 - c2 * sn0);
        dst[rowbase + 16 + lh] = (__bf16)(c1 * cc1 - c3 * sn1);
        dst[rowbase + 32 + lh] = (__bf16)(c2 * cc0 + c0 * sn0);
        dst[rowbase + 48 + lh] = (__bf16)(c3 * cc1 + c1 * sn1);
      }
    }
  }
}

// ---- kernel 4: flash attention, one 16-query tile per wave -----------------
__global__ __launch_bounds__(256) void attn_kernel(const __bf16* __restrict__ qb,
                                                   const __bf16* __restrict__ kb,
                                                   const __bf16* __restrict__ vt,
                                                   __bf16* __restrict__ ao) {
  __shared__ __bf16 plds[8][16 * 32];
  int lane = threadIdx.x & 31, wid = threadIdx.x >> 5;
  int wg = blockIdx.x * 8 + wid;
  int qt = wg % QTILES;
  int bh = wg / QTILES;  // 0..127
  int lh = lane & 15, lz = lane >> 4;
  int s0 = qt * 16;
  const float scale = 0.125f;  // 1/sqrt(64)

  const __bf16* qrow = qb + ((size_t)bh * SPAD + (s0 + lh)) * HD + lz * 16;
  v16bf aq0 = load_a16(qrow);
  v16bf aq1 = load_a16(qrow + 32);

  v8f o[4] = {{}, {}, {}, {}};
  float mrow[8], lrow[8];
#pragma unroll
  for (int r = 0; r < 8; ++r) { mrow[r] = -1e30f; lrow[r] = 0.f; }

  auto load_kf = [&](int kt, v16bf* kf) {
    int kk0 = kt * 32;
#pragma unroll
    for (int u = 0; u < 2; ++u) {
      const __bf16* krow =
          kb + ((size_t)bh * SPAD + (kk0 + u * 16 + lh)) * HD + lz * 16;
      kf[u * 2 + 0] = *(const v16bf*)(krow);
      kf[u * 2 + 1] = *(const v16bf*)(krow + 32);
    }
  };

  __bf16* pl = plds[wid];
  int ktmax = (s0 + 15) >> 5;
  v16bf kc[4], kn[4];
  load_kf(0, kc);
  for (int kt = 0; kt <= ktmax; ++kt) {
    int kk0 = kt * 32;
    v8f sc[2] = {{}, {}};
    sc[0] = wmma_bf16(aq0, kc[0], sc[0]);
    sc[0] = wmma_bf16(aq1, kc[1], sc[0]);
    sc[1] = wmma_bf16(aq0, kc[2], sc[1]);
    sc[1] = wmma_bf16(aq1, kc[3], sc[1]);
    // prefetch next K tile (last iteration reads zero padding; never consumed)
    load_kf(kt + 1, kn);
    // V frags for this tile, issued before the softmax VALU work hides them
    v16bf vf[4];
#pragma unroll
    for (int t = 0; t < 4; ++t)
      vf[t] = *(const v16bf*)(vt + ((size_t)bh * HD + t * 16 + lh) * SPAD +
                              kk0 + lz * 16);
    __builtin_amdgcn_sched_barrier(0);  // keep loads above the softmax chain
    // causal + bounds mask, per-row tile max
    float tmax[8];
#pragma unroll
    for (int r = 0; r < 8; ++r) {
      int qg = s0 + r + lz * 8;
      int c0g = kk0 + lh, c1g = kk0 + 16 + lh;
      float s0v = (c0g <= qg && c0g < S_) ? sc[0][r] * scale : -1e30f;
      float s1v = (c1g <= qg && c1g < S_) ? sc[1][r] * scale : -1e30f;
      sc[0][r] = s0v; sc[1][r] = s1v;
      float t = fmaxf(s0v, s1v);
      t = fmaxf(t, __shfl_xor(t, 1));
      t = fmaxf(t, __shfl_xor(t, 2));
      t = fmaxf(t, __shfl_xor(t, 4));
      t = fmaxf(t, __shfl_xor(t, 8));
      tmax[r] = t;
    }
    // online softmax update
#pragma unroll
    for (int r = 0; r < 8; ++r) {
      float mnew = fmaxf(mrow[r], tmax[r]);
      float corr = __expf(mrow[r] - mnew);
      mrow[r] = mnew;
      float p0 = __expf(sc[0][r] - mnew);
      float p1 = __expf(sc[1][r] - mnew);
      sc[0][r] = p0; sc[1][r] = p1;
      float rsum = p0 + p1;
      rsum += __shfl_xor(rsum, 1);
      rsum += __shfl_xor(rsum, 2);
      rsum += __shfl_xor(rsum, 4);
      rsum += __shfl_xor(rsum, 8);
      lrow[r] = lrow[r] * corr + rsum;
#pragma unroll
      for (int t = 0; t < 4; ++t) o[t][r] *= corr;
    }
    // P: D-layout -> LDS -> A-frag layout (wave-local, LDS is in-order)
#pragma unroll
    for (int r = 0; r < 8; ++r) {
      int row = r + lz * 8;
      pl[row * 32 + lh]      = (__bf16)sc[0][r];
      pl[row * 32 + 16 + lh] = (__bf16)sc[1][r];
    }
    __builtin_amdgcn_wave_barrier();
    v16bf pa = load_a16(pl + lh * 32 + lz * 16);
    __builtin_amdgcn_wave_barrier();
#pragma unroll
    for (int t = 0; t < 4; ++t) o[t] = wmma_bf16(pa, vf[t], o[t]);
#pragma unroll
    for (int i = 0; i < 4; ++i) kc[i] = kn[i];
  }

#pragma unroll
  for (int r = 0; r < 8; ++r) lrow[r] = 1.f / lrow[r];
  int bbat = bh >> 4, hh = bh & 15;
#pragma unroll
  for (int r = 0; r < 8; ++r) {
    int s = s0 + r + lz * 8;
    if (s >= S_) continue;
    size_t base = ((size_t)(bbat * S_ + s)) * DIM + hh * HD;
#pragma unroll
    for (int t = 0; t < 4; ++t)
      ao[base + t * 16 + lh] = (__bf16)(o[t][r] * lrow[r]);
  }
}

// ---- kernel 5: output projection GEMM (fp32 out + bias) --------------------
__global__ __launch_bounds__(256) void out_gemm(const __bf16* __restrict__ a,
                                                const __bf16* __restrict__ w,
                                                const float* __restrict__ bias,
                                                float* __restrict__ out) {
  int lane = threadIdx.x & 31, wid = threadIdx.x >> 5;
  int mt = blockIdx.x;
  int n0 = (blockIdx.y * 8 + wid) * 64;
  int lh = lane & 15, lz = lane >> 4;
  const __bf16* ap0 = a + (size_t)(mt * 32 + lh) * DIM + lz * 16;
  const __bf16* ap1 = ap0 + (size_t)16 * DIM;
  const __bf16* bp  = w + (size_t)(n0 + lh) * DIM + lz * 16;

  v8f acc[8] = {{}, {}, {}, {}, {}, {}, {}, {}};
  v16bf a0 = load_a16(ap0), a1 = load_a16(ap1);
  v16bf bf[4];
#pragma unroll
  for (int t = 0; t < 4; ++t) bf[t] = *(const v16bf*)(bp + (size_t)t * 16 * DIM);

#pragma unroll 4
  for (int k0 = 32; k0 < DIM; k0 += 32) {
    v16bf na0 = load_a16(ap0 + k0);
    v16bf na1 = load_a16(ap1 + k0);
    v16bf nb[4];
#pragma unroll
    for (int t = 0; t < 4; ++t)
      nb[t] = *(const v16bf*)(bp + (size_t)t * 16 * DIM + k0);
    __builtin_amdgcn_sched_barrier(0);  // keep prefetch ahead of the WMMAs
#pragma unroll
    for (int t = 0; t < 4; ++t) {
      acc[t]     = wmma_bf16(a0, bf[t], acc[t]);
      acc[4 + t] = wmma_bf16(a1, bf[t], acc[4 + t]);
    }
    a0 = na0; a1 = na1;
#pragma unroll
    for (int t = 0; t < 4; ++t) bf[t] = nb[t];
  }
#pragma unroll
  for (int t = 0; t < 4; ++t) {
    acc[t]     = wmma_bf16(a0, bf[t], acc[t]);
    acc[4 + t] = wmma_bf16(a1, bf[t], acc[4 + t]);
  }

  float bs[4];
#pragma unroll
  for (int t = 0; t < 4; ++t) bs[t] = bias[n0 + t * 16 + lh];
#pragma unroll
  for (int g = 0; g < 2; ++g) {
#pragma unroll
    for (int r = 0; r < 8; ++r) {
      int m = mt * 32 + g * 16 + r + lz * 8;
      if (m >= M_TOT) continue;
#pragma unroll
      for (int t = 0; t < 4; ++t)
        out[(size_t)m * DIM + n0 + t * 16 + lh] = acc[g * 4 + t][r] + bs[t];
    }
  }
}

// ---- host ------------------------------------------------------------------
extern "C" void kernel_launch(void* const* d_in, const int* in_sizes, int n_in,
                              void* d_out, int out_size, void* d_ws,
                              size_t ws_size, hipStream_t stream) {
  const float* x     = (const float*)d_in[0];
  const int*   pos   = (const int*)d_in[1];
  const float* theta = (const float*)d_in[2];
  const float* ln_w  = (const float*)d_in[3];
  const float* ln_b  = (const float*)d_in[4];
  const float* qkv_w = (const float*)d_in[5];
  const float* qkv_b = (const float*)d_in[6];
  const float* out_w = (const float*)d_in[7];
  const float* out_b = (const float*)d_in[8];
  float* out = (float*)d_out;

  char* ws = (char*)d_ws;
  size_t off = 0;
  __bf16* h   = (__bf16*)(ws + off); off += (size_t)M_PAD * DIM * 2;
  __bf16* wqk = (__bf16*)(ws + off); off += (size_t)3 * DIM * DIM * 2;
  __bf16* wo  = (__bf16*)(ws + off); off += (size_t)DIM * DIM * 2;
  __bf16* qB  = (__bf16*)(ws + off); off += (size_t)B_ * NH * SPAD * HD * 2;
  __bf16* kB  = (__bf16*)(ws + off); off += (size_t)B_ * NH * SPAD * HD * 2;
  __bf16* vT  = (__bf16*)(ws + off); off += (size_t)B_ * NH * SPAD * HD * 2;
  __bf16* ao  = (__bf16*)(ws + off); off += (size_t)M_PAD * DIM * 2;

  hipMemsetAsync(d_ws, 0, off, stream);  // zero padding deterministically

  cvt_bf16<<<3 * DIM * DIM / 1024, 256, 0, stream>>>(qkv_w, wqk, 3 * DIM * DIM);
  cvt_bf16<<<DIM * DIM / 1024, 256, 0, stream>>>(out_w, wo, DIM * DIM);
  ln_kernel<<<M_TOT, 256, 0, stream>>>(x, ln_w, ln_b, h);
  qkv_gemm<<<dim3(M_TILES2, 6), 256, 0, stream>>>(h, wqk, qkv_b, pos, theta, qB,
                                                  kB, vT);
  attn_kernel<<<(B_ * NH * QTILES) / 8, 256, 0, stream>>>(qB, kB, vT, ao);
  out_gemm<<<dim3(M_TILES2, 2), 256, 0, stream>>>(ao, wo, out_b, out);
}
